// Encoder_7997229105188
// MI455X (gfx1250) — compile-verified, matrix operand order checked by reading, not run on hardware
//
#include <hip/hip_runtime.h>
#include <math.h>

typedef __attribute__((ext_vector_type(16))) _Float16 v16h;
typedef __attribute__((ext_vector_type(8)))  float    v8f;

#define NN      8192
#define DIN     1024
#define DOUT    64
#define HID     512
#define NCAT    256
#define KF      4      /* k+1 for fixed hypergraphs */
#define KFU     5      /* k+1 for fused hypergraph  */
#define NORM_BLOCKS 1024

// ---------------------------------------------------------------------------
// Packed fragment layout for V_WMMA_F32_16X16X32_F16 (ISA 7.12.2, 16-bit A):
// lane = (m%16) + 16*((k%32 / 8) % 2); elem = (k%8) + 8*((k%32)/16)
// Fragment (16 halves) for (m-tile, k-tile, lane) stored contiguously (32B).
// B[K,N] packed identically with n taking the role of m.
// ---------------------------------------------------------------------------
static __device__ __forceinline__ size_t pack_off(int m, int k, int K) {
  int kt   = k >> 5, kr = k & 31;
  int lane = (m & 15) + (((kr >> 3) & 1) << 4);
  int e    = (kr & 7) + ((kr >> 4) << 3);
  return ((((size_t)(m >> 4)) * (size_t)(K >> 5) + kt) * 32 + lane) * 16 + e;
}

// ======================= norm / fusion weights =============================
__global__ void norm_partial(const float* __restrict__ ai, const float* __restrict__ ae,
                             const float* __restrict__ ag, float* __restrict__ partials) {
  size_t idx    = (size_t)blockIdx.x * blockDim.x + threadIdx.x;
  size_t stride = (size_t)gridDim.x * blockDim.x;
  const size_t total = (size_t)NN * NN;
  float s0 = 0.f, s1 = 0.f, s2 = 0.f;
  for (size_t i = idx; i < total; i += stride) {
    float x = ai[i], y = ae[i], z = ag[i];
    float f = (x + y + z) * (1.0f / 3.0f);
    float d0 = x - f, d1 = y - f, d2 = z - f;
    s0 += d0 * d0; s1 += d1 * d1; s2 += d2 * d2;
  }
  __shared__ float r0[256], r1[256], r2[256];
  r0[threadIdx.x] = s0; r1[threadIdx.x] = s1; r2[threadIdx.x] = s2;
  __syncthreads();
  for (int off = 128; off > 0; off >>= 1) {
    if ((int)threadIdx.x < off) {
      r0[threadIdx.x] += r0[threadIdx.x + off];
      r1[threadIdx.x] += r1[threadIdx.x + off];
      r2[threadIdx.x] += r2[threadIdx.x + off];
    }
    __syncthreads();
  }
  if (threadIdx.x == 0) {
    partials[blockIdx.x * 3 + 0] = r0[0];
    partials[blockIdx.x * 3 + 1] = r1[0];
    partials[blockIdx.x * 3 + 2] = r2[0];
  }
}

__global__ void norm_finalize(const float* __restrict__ partials, float* __restrict__ wv) {
  __shared__ float sw[3];
  if (threadIdx.x < 3) {
    float s = 0.f;
    for (int b = 0; b < NORM_BLOCKS; ++b) s += partials[b * 3 + threadIdx.x];
    sw[threadIdx.x] = 1.0f / (sqrtf(s) + 1e-8f);
  }
  __syncthreads();
  if (threadIdx.x == 0) {
    wv[0] = sw[0]; wv[1] = sw[1]; wv[2] = sw[2];
    wv[3] = 1.0f / (sw[0] + sw[1] + sw[2]);
  }
}

// ============================== top-k ======================================
// One 256-thread block per row; per-thread top-5 over a coalesced strided
// scan, then LDS merge with jax tie-break (equal value -> lower index).
__device__ __forceinline__ void topk_insert(float* v, int* id, float x, int c) {
  if (x > v[4]) {
    int pos = 4;
    if (x > v[3]) pos = 3;
    if (x > v[2]) pos = 2;
    if (x > v[1]) pos = 1;
    if (x > v[0]) pos = 0;
    for (int i = 4; i > pos; --i) { v[i] = v[i - 1]; id[i] = id[i - 1]; }
    v[pos] = x; id[pos] = c;
  }
}

__device__ __forceinline__ void topk_finish(float v[5], int id[5], int* __restrict__ nbr,
                                            int r, int kk) {
  __shared__ float sv[1280];
  __shared__ int   si[1280];
  #pragma unroll
  for (int i = 0; i < 5; ++i) { sv[threadIdx.x * 5 + i] = v[i]; si[threadIdx.x * 5 + i] = id[i]; }
  __syncthreads();
  if (threadIdx.x == 0) {
    for (int j = 0; j < kk; ++j) {
      float bv = -INFINITY; int bi = 0x7fffffff; int bp = 0;
      for (int t = 0; t < 1280; ++t) {
        float x = sv[t];
        if (x > bv || (x == bv && si[t] < bi)) { bv = x; bi = si[t]; bp = t; }
      }
      sv[bp] = -INFINITY;
      nbr[(size_t)r * kk + j] = bi;
    }
  }
}

__global__ void topk_plain(const float* __restrict__ adj, int* __restrict__ nbr, int kk) {
  int r = blockIdx.x;
  const float* row = adj + (size_t)r * NN;
  float v[5]; int id[5];
  #pragma unroll
  for (int i = 0; i < 5; ++i) { v[i] = -INFINITY; id[i] = 0x7fffffff; }
  for (int c = threadIdx.x; c < NN; c += 256) topk_insert(v, id, row[c], c);
  topk_finish(v, id, nbr, r, kk);
}

__global__ void topk_fused(const float* __restrict__ ai, const float* __restrict__ ae,
                           const float* __restrict__ ag, const float* __restrict__ wv,
                           int* __restrict__ nbr) {
  int r = blockIdx.x;
  const float* ri = ai + (size_t)r * NN;
  const float* re = ae + (size_t)r * NN;
  const float* rg = ag + (size_t)r * NN;
  float wi = wv[0], we = wv[1], wg = wv[2], inv = wv[3];
  float v[5]; int id[5];
  #pragma unroll
  for (int i = 0; i < 5; ++i) { v[i] = -INFINITY; id[i] = 0x7fffffff; }
  for (int c = threadIdx.x; c < NN; c += 256) {
    float x = (wi * ri[c] + we * re[c] + wg * rg[c]) * inv;
    topk_insert(v, id, x, c);
  }
  topk_finish(v, id, nbr, r, KFU);
}

// ====================== hyperedge degree (B) ===============================
__global__ void count_kernel(const int* __restrict__ nbr, int n, unsigned* __restrict__ cnt) {
  int i = blockIdx.x * blockDim.x + threadIdx.x;
  if (i < n) atomicAdd(&cnt[nbr[i]], 1u);
}
__global__ void binv_kernel(const unsigned* __restrict__ cnt, float* __restrict__ binv, int n) {
  int i = blockIdx.x * blockDim.x + threadIdx.x;
  if (i < n) binv[i] = cnt[i] ? 1.0f / (float)cnt[i] : 0.0f;
}

// ============================ packing ======================================
__global__ void pack_a(const float* __restrict__ A, _Float16* __restrict__ P, int M, int K) {
  size_t i = (size_t)blockIdx.x * blockDim.x + threadIdx.x;
  if (i >= (size_t)M * K) return;
  int m = (int)(i / K), k = (int)(i % K);
  P[pack_off(m, k, K)] = (_Float16)A[i];
}

__global__ void pack_bcat(const float* __restrict__ S0, const float* __restrict__ S1,
                          const float* __restrict__ S2, const float* __restrict__ S3,
                          _Float16* __restrict__ P) {
  size_t i = (size_t)blockIdx.x * blockDim.x + threadIdx.x;
  if (i >= (size_t)DIN * NCAT) return;
  int k = (int)(i / NCAT), n = (int)(i % NCAT);
  const float* S = (n < 64) ? S0 : (n < 128) ? S1 : (n < 192) ? S2 : S3;
  P[pack_off(n, k, DIN)] = (_Float16)S[(size_t)k * DOUT + (n & 63)];
}

__global__ void pack_b(const float* __restrict__ S, _Float16* __restrict__ P, int K, int N) {
  size_t i = (size_t)blockIdx.x * blockDim.x + threadIdx.x;
  if (i >= (size_t)K * N) return;
  int k = (int)(i / N), n = (int)(i % N);
  P[pack_off(n, k, K)] = (_Float16)S[i];
}

// ============================ WMMA GEMM ====================================
// One wave -> 16(M) x 64(N) tile: 4 accumulators, A fragment reused 4x.
// Straight-line k-loop: the compiler unrolls and emits clean clause-grouped
// global_load_b128 pairs per fragment; latency is hidden by occupancy
// (8 independent waves per block, small register footprint).
__global__ void wmma_gemm_kernel(const _Float16* __restrict__ Ap,
                                 const _Float16* __restrict__ Bp,
                                 float* __restrict__ C,
                                 const float* __restrict__ bias,
                                 int M, int N, int K) {
  const v16h* A = reinterpret_cast<const v16h*>(Ap);
  const v16h* B = reinterpret_cast<const v16h*>(Bp);
  int wid  = (blockIdx.x * blockDim.x + threadIdx.x) >> 5;
  int lane = threadIdx.x & 31;
  int tilesN4 = N >> 6;
  int tilesM  = M >> 4;
  if (wid >= tilesM * tilesN4) return;
  int mt  = wid / tilesN4;
  int nt4 = wid % tilesN4;
  int ks  = K >> 5;
  v8f acc0 = {}, acc1 = {}, acc2 = {}, acc3 = {};
  const v16h* Ab = A + ((size_t)mt * ks) * 32 + lane;
  const v16h* B0 = B + ((size_t)(nt4 * 4 + 0) * ks) * 32 + lane;
  const v16h* B1 = B + ((size_t)(nt4 * 4 + 1) * ks) * 32 + lane;
  const v16h* B2 = B + ((size_t)(nt4 * 4 + 2) * ks) * 32 + lane;
  const v16h* B3 = B + ((size_t)(nt4 * 4 + 3) * ks) * 32 + lane;
  for (int kt = 0; kt < ks; ++kt) {
    v16h a  = Ab[(size_t)kt * 32];
    v16h b0 = B0[(size_t)kt * 32];
    v16h b1 = B1[(size_t)kt * 32];
    v16h b2 = B2[(size_t)kt * 32];
    v16h b3 = B3[(size_t)kt * 32];
    acc0 = __builtin_amdgcn_wmma_f32_16x16x32_f16(false, a, false, b0, (short)0, acc0, false, false);
    acc1 = __builtin_amdgcn_wmma_f32_16x16x32_f16(false, a, false, b1, (short)0, acc1, false, false);
    acc2 = __builtin_amdgcn_wmma_f32_16x16x32_f16(false, a, false, b2, (short)0, acc2, false, false);
    acc3 = __builtin_amdgcn_wmma_f32_16x16x32_f16(false, a, false, b3, (short)0, acc3, false, false);
  }
  int row0 = (mt << 4) + ((lane >> 4) << 3);
  int ncol = lane & 15;
  int n0 = (nt4 * 4 + 0) * 16 + ncol;
  int n1 = (nt4 * 4 + 1) * 16 + ncol;
  int n2 = (nt4 * 4 + 2) * 16 + ncol;
  int n3 = (nt4 * 4 + 3) * 16 + ncol;
  #pragma unroll
  for (int r = 0; r < 8; ++r) {
    float* crow = C + (size_t)(row0 + r) * N;
    crow[n0] = acc0[r] + (bias ? bias[n0] : 0.0f);
    crow[n1] = acc1[r] + (bias ? bias[n1] : 0.0f);
    crow[n2] = acc2[r] + (bias ? bias[n2] : 0.0f);
    crow[n3] = acc3[r] + (bias ? bias[n3] : 0.0f);
  }
}

// ===================== hconv scatter / gather ==============================
__global__ void scatter3(const float* __restrict__ xt,
                         const int* __restrict__ ne, const int* __restrict__ ng,
                         const int* __restrict__ ni,
                         float* __restrict__ m0, float* __restrict__ m1,
                         float* __restrict__ m2) {
  int r = blockIdx.x, c = threadIdx.x;
  const float* xr = xt + (size_t)r * NCAT;
  float x0 = xr[c], x1 = xr[64 + c], x2 = xr[128 + c];
  #pragma unroll
  for (int j = 0; j < KF; ++j) {
    atomicAdd(&m0[(size_t)ne[r * KF + j] * 64 + c], x0);
    atomicAdd(&m1[(size_t)ng[r * KF + j] * 64 + c], x1);
    atomicAdd(&m2[(size_t)ni[r * KF + j] * 64 + c], x2);
  }
}

__global__ void combine_z(const float* __restrict__ xt,
                          const float* __restrict__ m0, const float* __restrict__ m1,
                          const float* __restrict__ m2,
                          const int* __restrict__ ne, const int* __restrict__ ng,
                          const int* __restrict__ ni,
                          const float* __restrict__ bve, const float* __restrict__ bvg,
                          const float* __restrict__ bvi,
                          const float* __restrict__ b1, const float* __restrict__ b2,
                          const float* __restrict__ b3,
                          float* __restrict__ z) {
  int r = blockIdx.x, c = threadIdx.x;
  float acc = xt[(size_t)r * NCAT + 192 + c] + b1[c] + b2[c] + b3[c];
  float s = 0.f;
  #pragma unroll
  for (int j = 0; j < KF; ++j) {
    int e0 = ne[r * KF + j]; s += bve[e0] * m0[(size_t)e0 * 64 + c];
    int e1 = ng[r * KF + j]; s += bvg[e1] * m1[(size_t)e1 * 64 + c];
    int e2 = ni[r * KF + j]; s += bvi[e2] * m2[(size_t)e2 * 64 + c];
  }
  z[(size_t)r * 64 + c] = acc + 0.25f * s;   /* Dinv = 1/(k+1) = 1/4 */
}

__global__ void scatter_f(const float* __restrict__ xt, const int* __restrict__ nf,
                          float* __restrict__ m) {
  int r = blockIdx.x, c = threadIdx.x;
  float x = xt[(size_t)r * 64 + c];
  #pragma unroll
  for (int j = 0; j < KFU; ++j)
    atomicAdd(&m[(size_t)nf[r * KFU + j] * 64 + c], x);
}

__global__ void g_kernel(const float* __restrict__ m, const int* __restrict__ nf,
                         const float* __restrict__ bvf, const float* __restrict__ b4,
                         float* __restrict__ g) {
  int r = blockIdx.x, c = threadIdx.x;
  float s = 0.f;
  #pragma unroll
  for (int j = 0; j < KFU; ++j) {
    int e = nf[r * KFU + j];
    s += bvf[e] * m[(size_t)e * 64 + c];
  }
  float val = 0.2f * s + b4[c];               /* Dinv = 1/5 */
  g[(size_t)r * 64 + c] = 1.0f / (1.0f + expf(-val));
}

// ====================== small dense kernels ================================
// t = relu(z @ Wl1 + bl1), written directly in packed-f16 fragment layout.
__global__ void t_kernel(const float* __restrict__ z, const float* __restrict__ Wl1,
                         const float* __restrict__ bl1, _Float16* __restrict__ tP) {
  __shared__ float zr[64];
  int r = blockIdx.x;
  if (threadIdx.x < 64) zr[threadIdx.x] = z[(size_t)r * 64 + threadIdx.x];
  __syncthreads();
  for (int c = threadIdx.x; c < HID; c += blockDim.x) {
    float acc = bl1[c];
    #pragma unroll 8
    for (int k = 0; k < 64; ++k) acc += zr[k] * Wl1[(size_t)k * HID + c];
    acc = fmaxf(acc, 0.f);
    tP[pack_off(r, c, HID)] = (_Float16)acc;
  }
}

__global__ void small_gemm64(const float* __restrict__ X, const float* __restrict__ W,
                             float* __restrict__ O) {   /* O[r][c] = sum_k X[r][k] W[k][c] */
  __shared__ float xr[64];
  int r = blockIdx.x, c = threadIdx.x;
  xr[c] = X[(size_t)r * 64 + c];
  __syncthreads();
  float acc = 0.f;
  #pragma unroll 8
  for (int k = 0; k < 64; ++k) acc += xr[k] * W[k * 64 + c];
  O[(size_t)r * 64 + c] = acc;
}

__global__ void small_gemm64T(const float* __restrict__ X, const float* __restrict__ W,
                              float* __restrict__ O) {  /* O[r][d] = sum_e X[r][e] W[d][e] */
  __shared__ float xr[64];
  int r = blockIdx.x, d = threadIdx.x;
  xr[d] = X[(size_t)r * 64 + d];
  __syncthreads();
  float acc = 0.f;
  #pragma unroll 8
  for (int e = 0; e < 64; ++e) acc += xr[e] * W[d * 64 + e];
  O[(size_t)r * 64 + d] = acc;
}

__global__ void rowdot(const float* __restrict__ z, const float* __restrict__ za,
                       const float* __restrict__ gW, const float* __restrict__ gaW,
                       const float* __restrict__ bd, float* __restrict__ ret,
                       float* __restrict__ reta) {
  int r = blockIdx.x * blockDim.x + threadIdx.x;
  if (r >= NN) return;
  float b = bd[0];
  float a0 = 0.f, a1 = 0.f, c0 = 0.f, c1 = 0.f;
  for (int d = 0; d < 64; ++d) {
    float ez  = fmaxf(z[(size_t)r * 64 + d], 0.f);
    float eza = fmaxf(za[(size_t)r * 64 + d], 0.f);
    float w0  = gW[(size_t)r * 64 + d];
    float w1  = gaW[(size_t)r * 64 + d];
    a0 += ez * w0;  a1 += eza * w0;
    c0 += eza * w1; c1 += ez * w1;
  }
  ret[r * 2 + 0]  = a0 + b;  ret[r * 2 + 1]  = a1 + b;
  reta[r * 2 + 0] = c0 + b;  reta[r * 2 + 1] = c1 + b;
}

// ============================= host side ===================================
extern "C" void kernel_launch(void* const* d_in, const int* in_sizes, int n_in,
                              void* d_out, int out_size, void* d_ws, size_t ws_size,
                              hipStream_t stream) {
  const float* feat    = (const float*)d_in[0];
  const float* feat_a  = (const float*)d_in[1];
  const float* adji    = (const float*)d_in[2];
  const float* adje    = (const float*)d_in[3];
  const float* adjg    = (const float*)d_in[4];
  const float* weight1 = (const float*)d_in[5];
  const float* W1  = (const float*)d_in[6];
  const float* b1  = (const float*)d_in[7];
  const float* W2  = (const float*)d_in[8];
  const float* b2  = (const float*)d_in[9];
  const float* W3  = (const float*)d_in[10];
  const float* b3  = (const float*)d_in[11];
  const float* W4  = (const float*)d_in[12];
  const float* b4  = (const float*)d_in[13];
  const float* Wl1 = (const float*)d_in[14];
  const float* bl1 = (const float*)d_in[15];
  const float* Wl2 = (const float*)d_in[16];
  const float* bl2 = (const float*)d_in[17];
  const float* Wd  = (const float*)d_in[18];
  const float* bd  = (const float*)d_in[19];
  (void)in_sizes; (void)n_in; (void)out_size; (void)ws_size;

  float* z_out = (float*)d_out;
  float* h_out = z_out + (size_t)NN * DOUT;
  float* ret   = h_out + (size_t)NN * DIN;
  float* ret_a = ret + (size_t)NN * 2;

  char* p = (char*)d_ws;
  auto alloc = [&](size_t bytes) -> char* {
    char* r = p; p += (bytes + 255) & ~(size_t)255; return r;
  };
  float*    partials = (float*)alloc(NORM_BLOCKS * 3 * sizeof(float));
  float*    wv    = (float*)alloc(4 * sizeof(float));
  int*      nbr_e = (int*)alloc((size_t)NN * KF * 4);
  int*      nbr_g = (int*)alloc((size_t)NN * KF * 4);
  int*      nbr_i = (int*)alloc((size_t)NN * KF * 4);
  int*      nbr_f = (int*)alloc((size_t)NN * KFU * 4);
  unsigned* cnt   = (unsigned*)alloc((size_t)4 * NN * 4);
  float*    binv  = (float*)alloc((size_t)4 * NN * 4);
  _Float16* WcatP = (_Float16*)alloc((size_t)DIN * NCAT * 2);
  _Float16* Wl2P  = (_Float16*)alloc((size_t)HID * DIN * 2);
  _Float16* featP = (_Float16*)alloc((size_t)NN * DIN * 2);     /* reused per branch */
  float*    xt    = (float*)alloc((size_t)NN * NCAT * 4);       /* reused per branch */
  float*    mbuf  = (float*)alloc((size_t)3 * NN * DOUT * 4);   /* m0|m1|m2, reused  */
  float*    m0 = mbuf, *m1 = mbuf + (size_t)NN * DOUT, *m2 = mbuf + (size_t)2 * NN * DOUT;
  float*    z_a   = (float*)alloc((size_t)NN * DOUT * 4);
  _Float16* tP    = (_Float16*)alloc((size_t)NN * HID * 2);
  float*    zW4b  = (float*)alloc((size_t)NN * DOUT * 4);       /* reused per branch */
  float*    g     = (float*)alloc((size_t)NN * DOUT * 4);
  float*    g_a   = (float*)alloc((size_t)NN * DOUT * 4);
  float*    gWb   = (float*)alloc((size_t)NN * DOUT * 4);
  float*    gaWb  = (float*)alloc((size_t)NN * DOUT * 4);

  // --- fusion weights (deterministic two-stage reduction) ---
  norm_partial<<<NORM_BLOCKS, 256, 0, stream>>>(adji, adje, adjg, partials);
  norm_finalize<<<1, 64, 0, stream>>>(partials, wv);

  // --- hypergraph construction (top-k) ---
  topk_plain<<<NN, 256, 0, stream>>>(adje, nbr_e, KF);
  topk_plain<<<NN, 256, 0, stream>>>(adjg, nbr_g, KF);
  topk_plain<<<NN, 256, 0, stream>>>(adji, nbr_i, KF);
  topk_fused<<<NN, 256, 0, stream>>>(adji, adje, adjg, wv, nbr_f);

  hipMemsetAsync(cnt, 0, (size_t)4 * NN * 4, stream);
  count_kernel<<<(NN * KF + 255) / 256, 256, 0, stream>>>(nbr_e, NN * KF, cnt);
  count_kernel<<<(NN * KF + 255) / 256, 256, 0, stream>>>(nbr_g, NN * KF, cnt + NN);
  count_kernel<<<(NN * KF + 255) / 256, 256, 0, stream>>>(nbr_i, NN * KF, cnt + 2 * NN);
  count_kernel<<<(NN * KFU + 255) / 256, 256, 0, stream>>>(nbr_f, NN * KFU, cnt + 3 * NN);
  binv_kernel<<<(4 * NN + 255) / 256, 256, 0, stream>>>(cnt, binv, 4 * NN);

  // --- pack weights for WMMA ---
  pack_bcat<<<((size_t)DIN * NCAT + 255) / 256, 256, 0, stream>>>(W1, W2, W3, weight1, WcatP);
  pack_b<<<((size_t)HID * DIN + 255) / 256, 256, 0, stream>>>(Wl2, Wl2P, HID, DIN);

  // --- branch(feat) -> z  and  branch(feat_a) -> z_a ---
  int gemm1_blocks = ((NN / 16) * (NCAT / 64) * 32 + 255) / 256;
  const float* xin[2]  = {feat, feat_a};
  float*       zdst[2] = {z_out, z_a};
  for (int br = 0; br < 2; ++br) {
    pack_a<<<((size_t)NN * DIN + 255) / 256, 256, 0, stream>>>(xin[br], featP, NN, DIN);
    wmma_gemm_kernel<<<gemm1_blocks, 256, 0, stream>>>(featP, WcatP, xt, nullptr, NN, NCAT, DIN);
    hipMemsetAsync(mbuf, 0, (size_t)3 * NN * DOUT * 4, stream);
    scatter3<<<NN, 64, 0, stream>>>(xt, nbr_e, nbr_g, nbr_i, m0, m1, m2);
    combine_z<<<NN, 64, 0, stream>>>(xt, m0, m1, m2, nbr_e, nbr_g, nbr_i,
                                     binv, binv + NN, binv + 2 * NN, b1, b2, b3, zdst[br]);
  }

  // --- h = relu(z @ Wl1 + bl1) @ Wl2 + bl2 ---
  t_kernel<<<NN, 128, 0, stream>>>(z_out, Wl1, bl1, tP);
  int gemmh_blocks = ((NN / 16) * (DIN / 64) * 32 + 255) / 256;
  wmma_gemm_kernel<<<gemmh_blocks, 256, 0, stream>>>(tP, Wl2P, h_out, bl2, NN, DIN, HID);

  // --- g = sigmoid(hconv(z, W4, b4, fused)); g_a likewise ---
  const float* zsrc[2] = {z_out, z_a};
  float*       gdst[2] = {g, g_a};
  for (int br = 0; br < 2; ++br) {
    small_gemm64<<<NN, 64, 0, stream>>>(zsrc[br], W4, zW4b);
    hipMemsetAsync(m0, 0, (size_t)NN * DOUT * 4, stream);
    scatter_f<<<NN, 64, 0, stream>>>(zW4b, nbr_f, m0);
    g_kernel<<<NN, 64, 0, stream>>>(m0, nbr_f, binv + 3 * NN, b4, gdst[br]);
  }

  // --- bilinear scores ---
  small_gemm64T<<<NN, 64, 0, stream>>>(g, Wd, gWb);
  small_gemm64T<<<NN, 64, 0, stream>>>(g_a, Wd, gaWb);
  rowdot<<<NN / 256, 256, 0, stream>>>(z_out, z_a, gWb, gaWb, bd, ret, ret_a);
}
